// SimpleGAT_88450556494573
// MI455X (gfx1250) — matrix operand order repaired and checked by reading, not compiled, not run on hardware
//
#include <hip/hip_runtime.h>
#include <hip/hip_bf16.h>
#include <stdint.h>

// ---------------------------------------------------------------------------
// Bipartite GATv2 for MI455X (gfx1250, wave32, WMMA).
//  - bf16 WMMA (v_wmma_f32_16x16x32_bf16) for node projections + final MLP
//  - activations pre-converted to K-padded bf16 so WMMA loops are pure
//    b128 loads + wmma (no guards, no in-loop cvt)
//  - fused edge-score pass (no 819MB EE materialization)
//  - segment softmax via ordered-int atomicMax + f32 atomicAdd (L2-resident)
// ---------------------------------------------------------------------------

typedef __attribute__((ext_vector_type(16))) __bf16 v16bf;
typedef __attribute__((ext_vector_type(8)))  float  v8f;

#define N_U 50000
#define N_M 10000
#define N_E 400000

__device__ __forceinline__ unsigned short f2bf(float f) {
  union { float f; unsigned u; } c; c.f = f;
  unsigned u = c.u + 0x7fffu + ((c.u >> 16) & 1u);   // round-to-nearest-even
  return (unsigned short)(u >> 16);
}

__device__ __forceinline__ unsigned enc_f32(float f) {
  union { float f; unsigned u; } c; c.f = f;
  return (c.u & 0x80000000u) ? ~c.u : (c.u | 0x80000000u);
}
__device__ __forceinline__ float dec_f32(unsigned k) {
  union { unsigned u; float f; } c;
  c.u = (k & 0x80000000u) ? (k & 0x7fffffffu) : ~k;
  return c.f;
}

union Frag { v16bf v; uint4 q[2]; unsigned short u[16]; };

// --------------------------- small utility kernels -------------------------

__global__ void k_fill_u32(unsigned* __restrict__ p, int n, unsigned v) {
  int i = blockIdx.x * 256 + threadIdx.x;
  if (i < n) p[i] = v;
}

__global__ void k_rows_bias(float* __restrict__ o, const float* __restrict__ b, int rows) {
  int i = blockIdx.x * 256 + threadIdx.x;
  if (i < rows * 512) o[i] = b[i & 511];
}

// W [K,N] f32 row-major  ->  Wt [N,Kp] bf16 row-major (K padded with zeros)
__global__ void k_wt_convert(const float* __restrict__ W, unsigned short* __restrict__ Wt,
                             int K, int N, int Kp) {
  int i = blockIdx.x * 256 + threadIdx.x;
  if (i >= N * Kp) return;
  int n = i / Kp, kp = i - n * Kp;
  Wt[i] = (kp < K) ? f2bf(W[(size_t)kp * N + n]) : (unsigned short)0;
}

// A [M,K] f32 row-major -> Ab [M,Kp] bf16 row-major (K padded with zeros)
__global__ void k_a_convert(const float* __restrict__ A, unsigned short* __restrict__ Ab,
                            int M, int K, int Kp) {
  int i = blockIdx.x * 256 + threadIdx.x;
  if (i >= M * Kp) return;
  int m = i / Kp, kp = i - m * Kp;
  Ab[i] = (kp < K) ? f2bf(A[(size_t)m * K + kp]) : (unsigned short)0;
}

// plain f32 -> bf16 elementwise
__global__ void k_h_convert(const float* __restrict__ in, unsigned short* __restrict__ o, int n) {
  int i = blockIdx.x * 256 + threadIdx.x;
  if (i < n) o[i] = f2bf(in[i]);
}

// --------------------------- edge feature MLP ------------------------------
// 38 -> 64 (relu) -> 32, one edge per thread, weights in LDS (broadcast reads)

__global__ __launch_bounds__(256) void k_edge_mlp(
    const float* __restrict__ ea,
    const float* __restrict__ w1, const float* __restrict__ b1,
    const float* __restrict__ w2, const float* __restrict__ b2,
    float* __restrict__ e32, int E) {
  __shared__ float s1[38 * 64];
  __shared__ float sb1[64];
  __shared__ float s2[64 * 32];
  __shared__ float sb2[32];
  for (int i = threadIdx.x; i < 38 * 64; i += 256) s1[i] = w1[i];
  for (int i = threadIdx.x; i < 64; i += 256) sb1[i] = b1[i];
  for (int i = threadIdx.x; i < 64 * 32; i += 256) s2[i] = w2[i];
  if (threadIdx.x < 32) sb2[threadIdx.x] = b2[threadIdx.x];
  __syncthreads();

  int e = blockIdx.x * 256 + threadIdx.x;
  if (e >= E) return;

  float x[38];
#pragma unroll
  for (int i = 0; i < 38; ++i) x[i] = ea[(size_t)e * 38 + i];

  float h[64];
#pragma unroll
  for (int j = 0; j < 64; ++j) {
    float a = sb1[j];
#pragma unroll
    for (int i = 0; i < 38; ++i) a += x[i] * s1[i * 64 + j];
    h[j] = fmaxf(a, 0.f);
  }
  for (int o = 0; o < 32; ++o) {
    float a = sb2[o];
#pragma unroll
    for (int j = 0; j < 64; ++j) a += h[j] * s2[j * 32 + o];
    e32[(size_t)e * 32 + o] = a;
  }
}

// --------------------------- bf16 WMMA GEMM --------------------------------
// C[M,N] = Ab[M,Kp](bf16) @ Wt[N,Kp](bf16)^T + bias.
// Block = 128 thr (4 waves); wave w computes rows [bx*64+w*16,+16) x 64 cols.
// Inner loop: 2x b128 (A) + 8x b128 (B) + 4 wmma, no guards.

__global__ __launch_bounds__(128) void k_gemm_bf16(
    const unsigned short* __restrict__ Ab, int M, int Kp,
    const unsigned short* __restrict__ Wt, const float* __restrict__ bias,
    float* __restrict__ C, int N) {
  const int lane = threadIdx.x & 31, wave = threadIdx.x >> 5;
  const int lo = lane & 15, hi = lane >> 4;
  const int m0 = blockIdx.x * 64 + wave * 16;
  const int n0 = blockIdx.y * 64;
  const int mrow = m0 + lo;
  const int mload = (mrow < M) ? mrow : (M - 1);   // clamp: loads in-bounds, store guarded
  const unsigned short* arow = Ab + (size_t)mload * Kp;

  v8f acc[4];
#pragma unroll
  for (int t = 0; t < 4; ++t) acc[t] = (v8f){0, 0, 0, 0, 0, 0, 0, 0};

  const int kiters = Kp >> 5;
  for (int kk = 0; kk < kiters; ++kk) {
    const int k0 = kk * 32;
    Frag af;                                       // elems 0..7: k=k0+hi*8+i ; 8..15: +16
    const unsigned short* ap = arow + k0 + hi * 8;
    af.q[0] = *(const uint4*)ap;
    af.q[1] = *(const uint4*)(ap + 16);
#pragma unroll
    for (int t = 0; t < 4; ++t) {
      Frag bf;                                     // lane: col n, elems 0..7: k=k0+hi*8+i ; 8..15: +16
      const unsigned short* br = Wt + (size_t)(n0 + t * 16 + lo) * Kp + k0 + hi * 8;
      bf.q[0] = *(const uint4*)br;
      bf.q[1] = *(const uint4*)(br + 16);
      acc[t] = __builtin_amdgcn_wmma_f32_16x16x32_bf16(
          false, af.v, false, bf.v, (short)0, acc[t], false, false);
    }
  }
#pragma unroll
  for (int t = 0; t < 4; ++t) {
    int n = n0 + t * 16 + lo;
    float bv = bias[n];
#pragma unroll
    for (int r = 0; r < 8; ++r) {
      int m = m0 + hi * 8 + r;
      if (m < M) C[(size_t)m * N + n] = acc[t][r] + bv;
    }
  }
}

// --------------------------- GAT score pass --------------------------------
// One wave per edge (grid-stride). ee = e32[e] @ we fused via LDS weights +
// wave shuffles. score[e,h] = sum_c leakyrelu(xl+xr+ee)*att ; atomicMax segmax.

__global__ __launch_bounds__(256) void k_gat_score(
    const float* __restrict__ XL, const int* __restrict__ idxL,
    const float* __restrict__ XR, const int* __restrict__ seg,
    const float* __restrict__ e32, const float* __restrict__ we,
    const float* __restrict__ att, float* __restrict__ score,
    unsigned* __restrict__ segmax, int E) {
  __shared__ float swe[32 * 512];                  // 64 KB of the 320 KB WGP LDS
  for (int i = threadIdx.x; i < 32 * 512; i += 256) swe[i] = we[i];
  __syncthreads();

  const int lane = threadIdx.x & 31, wave = threadIdx.x >> 5;
  const int stride = gridDim.x * 8;
  for (int e = blockIdx.x * 8 + wave; e < E; e += stride) {
    const int sI = idxL[e], dI = seg[e];
    const float ev = e32[(size_t)e * 32 + lane];

    float ee[16];
#pragma unroll
    for (int j = 0; j < 16; ++j) ee[j] = 0.f;
    for (int k = 0; k < 32; ++k) {
      float b = __shfl(ev, k, 32);
      const float* wk = &swe[k * 512];
#pragma unroll
      for (int j = 0; j < 16; ++j) ee[j] += b * wk[lane + 32 * j];
    }

    const float* xl = XL + (size_t)sI * 512;
    const float* xr = XR + (size_t)dI * 512;
    float s0 = 0.f, s1 = 0.f, s2 = 0.f, s3 = 0.f;
#pragma unroll
    for (int j = 0; j < 16; ++j) {
      int c = lane + 32 * j;                       // head = j>>2 for lane<32
      float z = xl[c] + xr[c] + ee[j];
      z = (z > 0.f) ? z : 0.2f * z;
      float p = z * att[c];
      if (j < 4) s0 += p; else if (j < 8) s1 += p; else if (j < 12) s2 += p; else s3 += p;
    }
#pragma unroll
    for (int off = 1; off < 32; off <<= 1) {
      s0 += __shfl_xor(s0, off, 32);
      s1 += __shfl_xor(s1, off, 32);
      s2 += __shfl_xor(s2, off, 32);
      s3 += __shfl_xor(s3, off, 32);
    }
    if (lane == 0) {
      size_t sb = (size_t)e * 4, db = (size_t)dI * 4;
      score[sb + 0] = s0; atomicMax(&segmax[db + 0], enc_f32(s0));
      score[sb + 1] = s1; atomicMax(&segmax[db + 1], enc_f32(s1));
      score[sb + 2] = s2; atomicMax(&segmax[db + 2], enc_f32(s2));
      score[sb + 3] = s3; atomicMax(&segmax[db + 3], enc_f32(s3));
    }
  }
}

// score -> exp(score - segmax), accumulate denom
__global__ void k_softmax_prep(float* __restrict__ score, const int* __restrict__ seg,
                               const unsigned* __restrict__ segmax,
                               float* __restrict__ denom, int E) {
  int i = blockIdx.x * 256 + threadIdx.x;
  if (i >= E * 4) return;
  int e = i >> 2, h = i & 3;
  int d = seg[e];
  float m = dec_f32(segmax[(size_t)d * 4 + h]);
  float ex = __expf(score[i] - m);
  score[i] = ex;
  atomicAdd(&denom[(size_t)d * 4 + h], ex);
}

// out[dst] += xl[src] * alpha  (one wave per edge, 16 channels per lane)
__global__ __launch_bounds__(256) void k_gat_agg(
    const float* __restrict__ XL, const int* __restrict__ idxL,
    const int* __restrict__ seg, const float* __restrict__ ex,
    const float* __restrict__ denom, float* __restrict__ out, int E) {
  const int lane = threadIdx.x & 31, wave = threadIdx.x >> 5;
  int e = blockIdx.x * 8 + wave;
  if (e >= E) return;
  const int sI = idxL[e], dI = seg[e];
  float a0 = ex[(size_t)e * 4 + 0] / denom[(size_t)dI * 4 + 0];
  float a1 = ex[(size_t)e * 4 + 1] / denom[(size_t)dI * 4 + 1];
  float a2 = ex[(size_t)e * 4 + 2] / denom[(size_t)dI * 4 + 2];
  float a3 = ex[(size_t)e * 4 + 3] / denom[(size_t)dI * 4 + 3];
  const float* xl = XL + (size_t)sI * 512;
  float* o = out + (size_t)dI * 512;
#pragma unroll
  for (int j = 0; j < 16; ++j) {
    int c = lane + 32 * j;
    float a = (j < 4) ? a0 : (j < 8) ? a1 : (j < 12) ? a2 : a3;
    atomicAdd(&o[c], xl[c] * a);
  }
}

// --------------------------- final MLP -------------------------------------
// out = relu([h_u[src], h_m[dst]] @ fc_w1 + b1) @ fc_w2 + b2
// One wave per 16 edges: 8 accumulator tiles (16x128), K=1024. A rows come
// from bf16 h buffers (L2-resident), pure b128 loads; B = bf16 W1t[128][1024].

__global__ __launch_bounds__(256) void k_final_mlp(
    const unsigned short* __restrict__ hub, const unsigned short* __restrict__ hmb,
    const int* __restrict__ src, const int* __restrict__ dst,
    const unsigned short* __restrict__ W1t, const float* __restrict__ b1,
    const float* __restrict__ w2, const float* __restrict__ b2,
    float* __restrict__ out, int E) {
  const int lane = threadIdx.x & 31, wave = threadIdx.x >> 5;
  const int lo = lane & 15, hi = lane >> 4;
  const int e0 = (blockIdx.x * 8 + wave) * 16;
  if (e0 >= E) return;

  const int eA = e0 + lo;                           // row this lane loads
  const int s = src[eA], d = dst[eA];
  const unsigned short* ru = hub + (size_t)s * 512;
  const unsigned short* rm = hmb + (size_t)d * 512 - 512;  // indexed with k in [512,1024)

  v8f acc[8];
#pragma unroll
  for (int t = 0; t < 8; ++t) acc[t] = (v8f){0, 0, 0, 0, 0, 0, 0, 0};

  for (int kk = 0; kk < 32; ++kk) {
    const int k0 = kk * 32;
    const int ka = k0 + hi * 8;                     // chunk never straddles 512
    const unsigned short* ba = (ka < 512) ? ru : rm;
    Frag af;
    af.q[0] = *(const uint4*)(ba + ka);
    af.q[1] = *(const uint4*)(ba + ka + 16);
#pragma unroll
    for (int t = 0; t < 8; ++t) {
      Frag bf;
      const unsigned short* br = W1t + (size_t)(t * 16 + lo) * 1024 + k0 + hi * 8;
      bf.q[0] = *(const uint4*)br;
      bf.q[1] = *(const uint4*)(br + 16);
      acc[t] = __builtin_amdgcn_wmma_f32_16x16x32_bf16(
          false, af.v, false, bf.v, (short)0, acc[t], false, false);
    }
  }

  // epilogue: relu + dot with fc_w2, reduce across the 16-lane half (same m set)
  float part[8];
#pragma unroll
  for (int r = 0; r < 8; ++r) part[r] = 0.f;
#pragma unroll
  for (int t = 0; t < 8; ++t) {
    int n = t * 16 + lo;
    float bv = b1[n], wv = w2[n];
#pragma unroll
    for (int r = 0; r < 8; ++r) {
      float v = acc[t][r] + bv;
      v = (v > 0.f) ? v : 0.f;
      part[r] += v * wv;
    }
  }
#pragma unroll
  for (int r = 0; r < 8; ++r) {
#pragma unroll
    for (int off = 1; off < 16; off <<= 1) part[r] += __shfl_xor(part[r], off, 32);
  }
  const float bias2 = b2[0];
  if (lo == 0) {
#pragma unroll
    for (int r = 0; r < 8; ++r) out[e0 + hi * 8 + r] = part[r] + bias2;
  }
}

// --------------------------- launcher --------------------------------------

extern "C" void kernel_launch(void* const* d_in, const int* in_sizes, int n_in,
                              void* d_out, int out_size, void* d_ws, size_t ws_size,
                              hipStream_t stream) {
  (void)in_sizes; (void)n_in; (void)out_size; (void)ws_size;

  const float* x_u    = (const float*)d_in[0];
  const float* x_m    = (const float*)d_in[1];
  const int*   src    = (const int*)d_in[2];
  const int*   dst    = (const int*)d_in[3];
  const float* e_attr = (const float*)d_in[4];
  const float* ep_w1  = (const float*)d_in[5];
  const float* ep_b1  = (const float*)d_in[6];
  const float* ep_w2  = (const float*)d_in[7];
  const float* ep_b2  = (const float*)d_in[8];
  const float* g1_wl  = (const float*)d_in[9];
  const float* g1_bl  = (const float*)d_in[10];
  const float* g1_wr  = (const float*)d_in[11];
  const float* g1_br  = (const float*)d_in[12];
  const float* g1_we  = (const float*)d_in[13];
  const float* g1_att = (const float*)d_in[14];
  const float* g1_bias= (const float*)d_in[15];
  const float* g2_wl  = (const float*)d_in[16];
  const float* g2_bl  = (const float*)d_in[17];
  const float* g2_wr  = (const float*)d_in[18];
  const float* g2_br  = (const float*)d_in[19];
  const float* g2_we  = (const float*)d_in[20];
  const float* g2_att = (const float*)d_in[21];
  const float* g2_bias= (const float*)d_in[22];
  const float* fc_w1  = (const float*)d_in[23];
  const float* fc_b1  = (const float*)d_in[24];
  const float* fc_w2  = (const float*)d_in[25];
  const float* fc_b2  = (const float*)d_in[26];
  float* out = (float*)d_out;

  const int E = N_E;
  char* wsb = (char*)d_ws;
  size_t off = 0;
  auto walloc = [&](size_t bytes) -> void* {
    void* p = wsb + off;
    off += (bytes + 255) & ~(size_t)255;
    return p;
  };

  float* e32 = (float*)walloc((size_t)E * 32 * 4);
  float* XL1 = (float*)walloc((size_t)N_U * 512 * 4);
  float* XR1 = (float*)walloc((size_t)N_M * 512 * 4);
  float* XL2 = (float*)walloc((size_t)N_M * 512 * 4);
  float* XR2 = (float*)walloc((size_t)N_U * 512 * 4);
  float* sc1 = (float*)walloc((size_t)E * 4 * 4);
  float* sc2 = (float*)walloc((size_t)E * 4 * 4);
  unsigned* mx1 = (unsigned*)walloc((size_t)N_M * 4 * 4);
  unsigned* mx2 = (unsigned*)walloc((size_t)N_U * 4 * 4);
  float* dn1 = (float*)walloc((size_t)N_M * 4 * 4);
  float* dn2 = (float*)walloc((size_t)N_U * 4 * 4);
  float* hm  = (float*)walloc((size_t)N_M * 512 * 4);
  float* hu  = (float*)walloc((size_t)N_U * 512 * 4);
  unsigned short* Wt1l = (unsigned short*)walloc((size_t)512 * 224 * 2);
  unsigned short* Wt1r = (unsigned short*)walloc((size_t)512 * 640 * 2);
  unsigned short* Wt2l = (unsigned short*)walloc((size_t)512 * 640 * 2);
  unsigned short* Wt2r = (unsigned short*)walloc((size_t)512 * 224 * 2);
  unsigned short* W1t  = (unsigned short*)walloc((size_t)128 * 1024 * 2);
  unsigned short* xub  = (unsigned short*)walloc((size_t)N_U * 224 * 2);
  unsigned short* xmb  = (unsigned short*)walloc((size_t)N_M * 640 * 2);
  unsigned short* hub  = (unsigned short*)walloc((size_t)N_U * 512 * 2);
  unsigned short* hmb  = (unsigned short*)walloc((size_t)N_M * 512 * 2);

  // 1) weight transpose + bf16 convert (K padded to multiple of 32)
  k_wt_convert<<<(512 * 224 + 255) / 256, 256, 0, stream>>>(g1_wl, Wt1l, 221, 512, 224);
  k_wt_convert<<<(512 * 640 + 255) / 256, 256, 0, stream>>>(g1_wr, Wt1r, 636, 512, 640);
  k_wt_convert<<<(512 * 640 + 255) / 256, 256, 0, stream>>>(g2_wl, Wt2l, 636, 512, 640);
  k_wt_convert<<<(512 * 224 + 255) / 256, 256, 0, stream>>>(g2_wr, Wt2r, 221, 512, 224);
  k_wt_convert<<<(128 * 1024 + 255) / 256, 256, 0, stream>>>(fc_w1, W1t, 1024, 128, 1024);

  // 1b) activation bf16 conversion (padded rows)
  k_a_convert<<<(N_U * 224 + 255) / 256, 256, 0, stream>>>(x_u, xub, N_U, 221, 224);
  k_a_convert<<<(N_M * 640 + 255) / 256, 256, 0, stream>>>(x_m, xmb, N_M, 636, 640);

  // 2) edge feature MLP
  k_edge_mlp<<<(E + 255) / 256, 256, 0, stream>>>(e_attr, ep_w1, ep_b1, ep_w2, ep_b2, e32, E);

  // 3) node projections (bf16 WMMA GEMMs)
  dim3 gU((N_U + 63) / 64, 8), gM((N_M + 63) / 64, 8);
  k_gemm_bf16<<<gU, 128, 0, stream>>>(xub, N_U, 224, Wt1l, g1_bl, XL1, 512);
  k_gemm_bf16<<<gM, 128, 0, stream>>>(xmb, N_M, 640, Wt1r, g1_br, XR1, 512);
  k_gemm_bf16<<<gM, 128, 0, stream>>>(xmb, N_M, 640, Wt2l, g2_bl, XL2, 512);
  k_gemm_bf16<<<gU, 128, 0, stream>>>(xub, N_U, 224, Wt2r, g2_br, XR2, 512);

  // 4) init softmax stats + output accumulators (h = bias rows)
  k_fill_u32<<<(N_M * 4 + 255) / 256, 256, 0, stream>>>(mx1, N_M * 4, 0u);
  k_fill_u32<<<(N_U * 4 + 255) / 256, 256, 0, stream>>>(mx2, N_U * 4, 0u);
  k_fill_u32<<<(N_M * 4 + 255) / 256, 256, 0, stream>>>((unsigned*)dn1, N_M * 4, 0u);
  k_fill_u32<<<(N_U * 4 + 255) / 256, 256, 0, stream>>>((unsigned*)dn2, N_U * 4, 0u);
  k_rows_bias<<<(N_M * 512 + 255) / 256, 256, 0, stream>>>(hm, g1_bias, N_M);
  k_rows_bias<<<(N_U * 512 + 255) / 256, 256, 0, stream>>>(hu, g2_bias, N_U);

  // 5) edge scores + segment max   (dir1: dst=movie ; dir2: dst=user)
  k_gat_score<<<1024, 256, 0, stream>>>(XL1, src, XR1, dst, e32, g1_we, g1_att, sc1, mx1, E);
  k_gat_score<<<1024, 256, 0, stream>>>(XL2, dst, XR2, src, e32, g2_we, g2_att, sc2, mx2, E);

  // 6) exp + segment sum
  k_softmax_prep<<<(E * 4 + 255) / 256, 256, 0, stream>>>(sc1, dst, mx1, dn1, E);
  k_softmax_prep<<<(E * 4 + 255) / 256, 256, 0, stream>>>(sc2, src, mx2, dn2, E);

  // 7) alpha-weighted scatter aggregation
  k_gat_agg<<<E / 8, 256, 0, stream>>>(XL1, src, dst, sc1, dn1, hm, E);
  k_gat_agg<<<E / 8, 256, 0, stream>>>(XL2, dst, src, sc2, dn2, hu, E);

  // 7b) h -> bf16 for the final MLP gather (halves gather bytes, no in-loop cvt)
  k_h_convert<<<(N_M * 512 + 255) / 256, 256, 0, stream>>>(hm, hmb, N_M * 512);
  k_h_convert<<<(N_U * 512 + 255) / 256, 256, 0, stream>>>(hu, hub, N_U * 512);

  // 8) final per-edge MLP (WMMA) -> d_out
  k_final_mlp<<<E / 16 / 8, 256, 0, stream>>>(hub, hmb, src, dst, W1t, fc_b1, fc_w2, fc_b2, out, E);
}
